// ActorWrapper_50964081935486
// MI455X (gfx1250) — matrix operand order repaired and checked by reading, not compiled, not run on hardware
//
#include <hip/hip_runtime.h>

// ---------------------------------------------------------------------------
// MI455X (gfx1250) fused gather + 2-layer MLP heads using v_wmma_f32_16x16x32_f16.
// wave32; 256 threads/block = 8 waves; 64 gathered rows per block.
// ---------------------------------------------------------------------------

#define LDS_STRIDE 264   // 256 + 8 f16 pad (16B) -> keeps 16B alignment, breaks bank conflicts
#define M_BLOCK    64

typedef _Float16 half4_t  __attribute__((ext_vector_type(4)));
typedef _Float16 half8_t  __attribute__((ext_vector_type(8)));
typedef _Float16 v16h     __attribute__((ext_vector_type(16)));
typedef float    v8f      __attribute__((ext_vector_type(8)));
typedef float    float4_t __attribute__((ext_vector_type(4)));

__device__ __forceinline__ v8f wmma_f16(v16h a, v16h b, v8f c) {
  // (neg_a, A, neg_b, B, c_mod, C, reuse_a, reuse_b)
  return __builtin_amdgcn_wmma_f32_16x16x32_f16(false, a, false, b, (short)0, c,
                                                false, false);
}

__device__ __forceinline__ v16h cat8(half8_t lo, half8_t hi) {
  return __builtin_shufflevector(lo, hi, 0, 1, 2, 3, 4, 5, 6, 7,
                                 8, 9, 10, 11, 12, 13, 14, 15);
}

// A-matrix 16x32 f16 fragment (CDNA5 layout):
//   lane row m = lane&15; lanes 0-15 hold K {0..7,16..23}, lanes 16-31 {8..15,24..31}
//   -> two contiguous 16B chunks per lane.
__device__ __forceinline__ v16h load_a_frag(const _Float16* __restrict__ x,
                                            int mt, int k0, int lane) {
  const _Float16* base = x + (mt * 16 + (lane & 15)) * LDS_STRIDE + k0 + (lane >> 4) * 8;
  half8_t lo = *(const half8_t*)(base);
  half8_t hi = *(const half8_t*)(base + 16);
  return cat8(lo, hi);
}

// B-matrix 32x16 f16 fragment: column n = n0 + (lane&15); lanes 0-15 K {0..7,16..23},
// lanes 16-31 K {8..15,24..31} -> two contiguous 16B chunks per lane. Weights are stored
// row-major [n][k] (i.e. w.T applied on the fly), 256 f16 per row.
__device__ __forceinline__ v16h load_b_frag(const _Float16* __restrict__ w,
                                            int n0, int k0, int lane) {
  const _Float16* base = w + (size_t)(n0 + (lane & 15)) * 256 + k0 + (lane >> 4) * 8;
  half8_t lo = *(const half8_t*)(base);
  half8_t hi = *(const half8_t*)(base + 16);
  return cat8(lo, hi);
}

// ---------------------------------------------------------------------------
// Fused head kernel. IS_BOND=false: 105-wide output head (stem).
//                    IS_BOND=true : 1-wide head + pair mean (bond).
// ---------------------------------------------------------------------------
template <bool IS_BOND>
__global__ __launch_bounds__(256)
void mlp_head_kernel(const float* __restrict__ per_atom,
                     const _Float16* __restrict__ w1, const float* __restrict__ b1,
                     const _Float16* __restrict__ w2, const float* __restrict__ b2,
                     const int* __restrict__ slices,
                     const int* __restrict__ batch_idx,
                     const int* __restrict__ atom_idx,
                     float* __restrict__ out) {
  __shared__ __align__(16) _Float16 xbuf[M_BLOCK * LDS_STRIDE];
  __shared__ __align__(16) _Float16 hbuf[M_BLOCK * LDS_STRIDE];
  __shared__ float bp[M_BLOCK];

  const int tid  = threadIdx.x;
  const int lane = tid & 31;
  const int wave = tid >> 5;
  const int mt   = wave & 3;   // M sub-tile (4 x 16 rows)
  const int ng   = wave >> 2;  // N group (2)
  const long long row_base = (long long)blockIdx.x * M_BLOCK;

  // ---- Stage 1: gather 64 rows, f32 -> f16, into LDS (coalesced 16B loads) --
  {
    const int sub = tid & 63;  // 64 threads cover one 256-float row (4 floats each)
    const int rg  = tid >> 6;  // 4 rows in flight
    for (int g = 0; g < 16; ++g) {
      const int lr = g * 4 + rg;
      const long long grow = row_base + lr;
      int gi;
      if (IS_BOND) gi = slices[batch_idx[grow >> 1]] + atom_idx[grow];
      else         gi = slices[batch_idx[grow]]      + atom_idx[grow];
      float4_t v = *(const float4_t*)(per_atom + (size_t)gi * 256 + sub * 4);
      half4_t h = {(_Float16)v[0], (_Float16)v[1], (_Float16)v[2], (_Float16)v[3]};
      *(half4_t*)&xbuf[lr * LDS_STRIDE + sub * 4] = h;
    }
  }
  __syncthreads();

  // ---- Stage 2: h = LeakyReLU(x @ w1^T + b1) -> LDS f16 --------------------
  for (int nt = ng * 8; nt < ng * 8 + 8; ++nt) {  // 16 n-tiles split over 2 groups
    v8f acc = {};
#pragma unroll
    for (int kk = 0; kk < 8; ++kk) {
      v16h a = load_a_frag(xbuf, mt, kk * 32, lane);
      v16h b = load_b_frag(w1, nt * 16, kk * 32, lane);
      acc = wmma_f16(a, b, acc);
    }
    // D layout: lane col n = lane&15; VGPR r -> row r + (lane>=16 ? 8 : 0)
    const int n = nt * 16 + (lane & 15);
    const float bias = b1[n];
    const int mb = mt * 16 + (lane >> 4) * 8;
#pragma unroll
    for (int r = 0; r < 8; ++r) {
      float v = acc[r] + bias;
      v = (v >= 0.0f) ? v : 0.01f * v;  // LeakyReLU
      hbuf[(mb + r) * LDS_STRIDE + n] = (_Float16)v;
    }
  }
  __syncthreads();

  // ---- Stage 3: out = h @ w2^T + b2 ----------------------------------------
  if (IS_BOND) {
    if (ng == 0) {  // wave-uniform: EXEC stays all-ones inside WMMA
      v8f acc = {};
#pragma unroll
      for (int kk = 0; kk < 8; ++kk) {
        v16h a = load_a_frag(hbuf, mt, kk * 32, lane);
        v16h b = load_b_frag(w2, 0, kk * 32, lane);
        acc = wmma_f16(a, b, acc);
      }
      if ((lane & 15) == 0) {  // only column 0 is a real output
        const float bias = b2[0];
        const int mb = mt * 16 + (lane >> 4) * 8;
#pragma unroll
        for (int r = 0; r < 8; ++r) bp[mb + r] = acc[r] + bias;
      }
    }
    __syncthreads();
    // pair mean: rows 2i, 2i+1 are endpoints of bond i (block start is even)
    if (tid < M_BLOCK / 2)
      out[row_base / 2 + tid] = 0.5f * (bp[2 * tid] + bp[2 * tid + 1]);
  } else {
    const int nt_lo = ng * 4;
    const int nt_hi = (ng == 0) ? 4 : 7;  // 7 tiles cover 105 (padded to 112)
    for (int nt = nt_lo; nt < nt_hi; ++nt) {
      v8f acc = {};
#pragma unroll
      for (int kk = 0; kk < 8; ++kk) {
        v16h a = load_a_frag(hbuf, mt, kk * 32, lane);
        v16h b = load_b_frag(w2, nt * 16, kk * 32, lane);
        acc = wmma_f16(a, b, acc);
      }
      const int n = nt * 16 + (lane & 15);
      if (n < 105) {
        const float bias = b2[n];
        const int mb = mt * 16 + (lane >> 4) * 8;
#pragma unroll
        for (int r = 0; r < 8; ++r)
          out[(size_t)(row_base + mb + r) * 105 + n] = acc[r] + bias;
      }
    }
  }
}

// ---------------------------------------------------------------------------
// One-time (per launch) weight conversion: f32 -> f16, with zero padding of the
// layer-2 output dims (105 -> 112, 1 -> 16) so GEMM2 tiles never read OOB.
// Workspace f16 layout: w_s1[65536] | w_s2p[112*256] | w_b1[65536] | w_b2p[16*256]
// ---------------------------------------------------------------------------
#define WS_S1 0
#define WS_S2 65536
#define WS_B1 94208
#define WS_B2 159744
#define WS_END 163840

__global__ __launch_bounds__(256)
void prep_weights_kernel(const float* __restrict__ ws1, const float* __restrict__ ws2,
                         const float* __restrict__ wb1, const float* __restrict__ wb2,
                         _Float16* __restrict__ wout) {
  const int i = blockIdx.x * 256 + threadIdx.x;
  if (i < WS_S2) {
    wout[i] = (_Float16)ws1[i];
  } else if (i < WS_B1) {
    const int j = i - WS_S2, n = j >> 8, k = j & 255;
    wout[i] = (n < 105) ? (_Float16)ws2[n * 256 + k] : (_Float16)0.0f;
  } else if (i < WS_B2) {
    wout[i] = (_Float16)wb1[i - WS_B1];
  } else if (i < WS_END) {
    const int j = i - WS_B2, n = j >> 8, k = j & 255;
    wout[i] = (n < 1) ? (_Float16)wb2[k] : (_Float16)0.0f;
  }
}

__global__ __launch_bounds__(256)
void copy_f32_kernel(const float* __restrict__ in, float* __restrict__ out, int n) {
  const int i = blockIdx.x * 256 + threadIdx.x;
  if (i < n) out[i] = in[i];
}

// ---------------------------------------------------------------------------
extern "C" void kernel_launch(void* const* d_in, const int* in_sizes, int n_in,
                              void* d_out, int out_size, void* d_ws, size_t ws_size,
                              hipStream_t stream) {
  const float* per_atom    = (const float*)d_in[0];
  const float* per_mol     = (const float*)d_in[1];
  const float* w_s1        = (const float*)d_in[2];
  const float* b_s1        = (const float*)d_in[3];
  const float* w_s2        = (const float*)d_in[4];
  const float* b_s2        = (const float*)d_in[5];
  const float* w_b1        = (const float*)d_in[6];
  const float* b_b1        = (const float*)d_in[7];
  const float* w_b2        = (const float*)d_in[8];
  const float* b_b2        = (const float*)d_in[9];
  const int*   slices      = (const int*)d_in[10];
  const int*   stems_batch = (const int*)d_in[11];
  const int*   stems       = (const int*)d_in[12];
  const int*   bonds       = (const int*)d_in[13];
  const int*   bonds_batch = (const int*)d_in[14];

  const int n_stems = in_sizes[11];  // 131072
  const int n_bonds = in_sizes[14];  // 516096
  const int B       = in_sizes[1];   // 8192

  _Float16* wbuf = (_Float16*)d_ws;

  float* out_stem = (float*)d_out;                       // [n_stems, 105]
  float* out_mol  = out_stem + (size_t)n_stems * 105;    // [B]
  float* out_bond = out_mol + B;                         // [n_bonds]

  hipLaunchKernelGGL(prep_weights_kernel, dim3(WS_END / 256), dim3(256), 0, stream,
                     w_s1, w_s2, w_b1, w_b2, wbuf);
  hipLaunchKernelGGL(copy_f32_kernel, dim3((B + 255) / 256), dim3(256), 0, stream,
                     per_mol, out_mol, B);
  hipLaunchKernelGGL((mlp_head_kernel<false>), dim3(n_stems / M_BLOCK), dim3(256), 0,
                     stream, per_atom, wbuf + WS_S1, b_s1, wbuf + WS_S2, b_s2,
                     slices, stems_batch, stems, out_stem);
  hipLaunchKernelGGL((mlp_head_kernel<true>), dim3((2 * n_bonds) / M_BLOCK), dim3(256), 0,
                     stream, per_atom, wbuf + WS_B1, b_b1, wbuf + WS_B2, b_b2,
                     slices, bonds_batch, bonds, out_bond);
}
